// GatedAttention_26792005992798
// MI455X (gfx1250) — compile-verified, matrix operand order checked by reading, not run on hardware
//
#include <hip/hip_runtime.h>
#include <math.h>
#include <stdint.h>

// ---------------- problem constants ----------------
constexpr int H    = 16;
constexpr int KV   = 4;
constexpr int D    = 128;
constexpr int HID  = 2048;
constexpr int Bb   = 2;
constexpr int Ss   = 2048;
constexpr int Mrows = Bb * Ss;                 // 4096
constexpr int NQKV  = 2 * H * D + 2 * KV * D;  // 5120
constexpr float EPSC = 1e-6f;

// ---------------- types ----------------
typedef __bf16 bf16;
typedef __attribute__((ext_vector_type(4)))  bf16  v4bf;
typedef __attribute__((ext_vector_type(8)))  bf16  v8bf;
typedef __attribute__((ext_vector_type(16))) bf16  v16bf;
typedef __attribute__((ext_vector_type(8)))  float v8f;

__device__ inline v16bf cat8(v8bf lo, v8bf hi) {
  return __builtin_shufflevector(lo, hi, 0,1,2,3,4,5,6,7,8,9,10,11,12,13,14,15);
}
// Load one 16-bit A/B fragment's per-lane data: two contiguous 16B chunks
// (halves koff..koff+7 and koff+16..koff+23 of a 32-element K run).
__device__ inline v16bf load_frag(const bf16* p) {
  v8bf lo = *(const v8bf*)(p);
  v8bf hi = *(const v8bf*)(p + 16);
  return cat8(lo, hi);
}

// ---- CDNA5 async global->LDS copy (ASYNCcnt path, bypasses VGPRs) ----
__device__ inline void async_ld_b128(uint32_t lds_off, const void* gaddr) {
  asm volatile("global_load_async_to_lds_b128 %0, %1, off"
               :: "v"(lds_off), "v"((unsigned long long)(uintptr_t)gaddr)
               : "memory");
}
__device__ inline void wait_async0() {
  asm volatile("s_wait_asynccnt 0x0" ::: "memory");
}
__device__ inline uint32_t lds_off_of(const void* p) {
  return (uint32_t)(uintptr_t)p;   // generic LDS addr: [31:0] is the LDS offset
}

__device__ inline float wave_sum32(float v) {
  #pragma unroll
  for (int m = 1; m < 32; m <<= 1) v += __shfl_xor(v, m, 32);
  return v;
}
__device__ inline float grp16_max(float v) {
  #pragma unroll
  for (int m = 1; m < 16; m <<= 1) v = fmaxf(v, __shfl_xor(v, m, 32));
  return v;
}
__device__ inline float grp16_sum(float v) {
  #pragma unroll
  for (int m = 1; m < 16; m <<= 1) v += __shfl_xor(v, m, 32);
  return v;
}

// ---------------- fp32 -> bf16 conversion ----------------
__global__ void ga_cvt_bf16(const float* __restrict__ in, bf16* __restrict__ out, size_t n) {
  size_t i = ((size_t)blockIdx.x * blockDim.x + threadIdx.x) * 4;
  if (i >= n) return;
  float4 f = *(const float4*)(in + i);
  v4bf o;
  o[0] = (bf16)f.x; o[1] = (bf16)f.y; o[2] = (bf16)f.z; o[3] = (bf16)f.w;
  *(v4bf*)(out + i) = o;
}

// ---------------- WMMA bf16 GEMM: C[M,N] = A[M,K] * W[N,K]^T ----------------
// 256x128 block tile, 8 waves (4x2), each wave 64(M) x 64(N), K step 32.
// Double-buffered async global->LDS staging; 16 WMMA per 8 fragment loads.
constexpr int TK = 32;
constexpr int TM = 256;
constexpr int TN = 128;
__global__ __launch_bounds__(256)
void ga_gemm_bf16(const bf16* __restrict__ A, const bf16* __restrict__ W,
                  float* __restrict__ C, int Mdim, int Ndim, int K) {
  __shared__ bf16 lA[2][TM * TK];   // 2 x 16KB
  __shared__ bf16 lB[2][TN * TK];   // 2 x 8KB
  const int nblk = Ndim / TN;
  const int m0 = (blockIdx.x / nblk) * TM;
  const int n0 = (blockIdx.x % nblk) * TN;
  const int tid  = threadIdx.x;
  const int wave = tid >> 5, lane = tid & 31;
  const int wy = wave >> 1, wx = wave & 1;         // 4x2 wave grid
  const int r    = lane & 15;
  const int koff = (lane >> 4) * 8;                // half-group K offset

  // staging map: A -> 1 row (64B = 4 x b128) per thread; B -> 32B per thread
  const int rowA = tid;                            // 0..255
  const int rowB = tid >> 1;                       // 0..127
  const int selB = (tid & 1) * 16;                 // halves
  const bf16* srcA = A + (size_t)(m0 + rowA) * K;
  const bf16* srcB = W + (size_t)(n0 + rowB) * K + selB;
  const uint32_t dA0 = lds_off_of(&lA[0][rowA * TK]);
  const uint32_t dB0 = lds_off_of(&lB[0][rowB * TK + selB]);
  const uint32_t strA = lds_off_of(&lA[1][0]) - lds_off_of(&lA[0][0]);
  const uint32_t strB = lds_off_of(&lB[1][0]) - lds_off_of(&lB[0][0]);

  auto stage = [&](int buf, int k0) {
    const bf16* a = srcA + k0;
    const bf16* b = srcB + k0;
    const uint32_t da = dA0 + buf * strA;
    const uint32_t db = dB0 + buf * strB;
    async_ld_b128(da,      a);
    async_ld_b128(da + 16, a + 8);
    async_ld_b128(da + 32, a + 16);
    async_ld_b128(da + 48, a + 24);
    async_ld_b128(db,      b);
    async_ld_b128(db + 16, b + 8);
  };

  v8f acc[4][4] = {};

  // prologue: stage tile 0 into buffer 0
  stage(0, 0);
  wait_async0();
  __syncthreads();

  for (int k0 = 0; k0 < K; k0 += TK) {
    const int cur = (k0 / TK) & 1;
    if (k0 + TK < K) stage(cur ^ 1, k0 + TK);   // overlap next-tile copy
    v16bf af[4], bfv[4];
    #pragma unroll
    for (int i = 0; i < 4; ++i)
      af[i] = load_frag(lA[cur] + (wy * 64 + i * 16 + r) * TK + koff);
    #pragma unroll
    for (int j = 0; j < 4; ++j)
      bfv[j] = load_frag(lB[cur] + (wx * 64 + j * 16 + r) * TK + koff);
    #pragma unroll
    for (int i = 0; i < 4; ++i)
      #pragma unroll
      for (int j = 0; j < 4; ++j)
        acc[i][j] = __builtin_amdgcn_wmma_f32_16x16x32_bf16(
            false, af[i], false, bfv[j], (short)0, acc[i][j], false, false);
    // next buffer's async writes complete + all waves done reading cur
    wait_async0();
    __syncthreads();
  }
  // C-layout store: VGPR g, lanes 0-15 -> row g, lanes 16-31 -> row g+8, col = lane&15
  const int rsel = (lane >> 4) * 8;
  #pragma unroll
  for (int i = 0; i < 4; ++i)
    #pragma unroll
    for (int j = 0; j < 4; ++j)
      #pragma unroll
      for (int g = 0; g < 8; ++g)
        C[(size_t)(m0 + wy * 64 + i * 16 + rsel + g) * Ndim +
          (n0 + wx * 64 + j * 16 + r)] = acc[i][j][g];
}

// ---------------- LayerNorm + RoPE, pack Q/K (row-major D) and V^T [D,S] ----------------
__global__ __launch_bounds__(256)
void ga_normrope(const float* __restrict__ qkv,
                 const float* __restrict__ qw, const float* __restrict__ qb,
                 const float* __restrict__ kw, const float* __restrict__ kb,
                 bf16* __restrict__ Qo,   // [B,H,S,D]
                 bf16* __restrict__ Ko,   // [B,KV,S,D]
                 bf16* __restrict__ VTo)  // [B,KV,D,S]
{
  const int ms = blockIdx.x;            // row of qkv
  const int b = ms / Ss, s = ms % Ss;
  const int wave = threadIdx.x >> 5, lane = threadIdx.x & 31;
  const float l2b = 19.931568569324174f / 64.0f;   // log2(1e6)/64

  for (int t = wave; t < 24; t += 8) {  // 24 head-tasks: 16 q + 4 k + 4 v
    int colbase, isv = 0;
    const float *nw = qw, *nb = qb;
    bf16* dst = nullptr;
    if (t < H) {                       // q head
      colbase = t * D;
      dst = Qo + (((size_t)b * H + t) * Ss + s) * D;
    } else if (t < H + KV) {           // k head
      colbase = 2 * H * D + (t - H) * D;
      nw = kw; nb = kb;
      dst = Ko + (((size_t)b * KV + (t - H)) * Ss + s) * D;
    } else {                           // v head
      colbase = 2 * H * D + KV * D + (t - H - KV) * D;
      isv = 1;
    }
    const int d0 = lane * 4;
    float4 xv = *(const float4*)(qkv + (size_t)ms * NQKV + colbase + d0);
    if (isv) {
      const int vh = t - H - KV;
      bf16* vt = VTo + (((size_t)b * KV + vh) * D) * Ss + s;
      vt[(size_t)(d0 + 0) * Ss] = (bf16)xv.x;
      vt[(size_t)(d0 + 1) * Ss] = (bf16)xv.y;
      vt[(size_t)(d0 + 2) * Ss] = (bf16)xv.z;
      vt[(size_t)(d0 + 3) * Ss] = (bf16)xv.w;
      continue;
    }
    // layernorm over D=128 (4 elems/lane, wave32 reduce)
    float mu = wave_sum32(xv.x + xv.y + xv.z + xv.w) * (1.0f / D);
    float c0 = xv.x - mu, c1 = xv.y - mu, c2 = xv.z - mu, c3 = xv.w - mu;
    float var = wave_sum32(c0*c0 + c1*c1 + c2*c2 + c3*c3) * (1.0f / D);
    float rs = rsqrtf(var + EPSC);
    float4 wv = *(const float4*)(nw + d0);
    float4 bv = *(const float4*)(nb + d0);
    float y0 = c0 * rs * wv.x + bv.x, y1 = c1 * rs * wv.y + bv.y;
    float y2 = c2 * rs * wv.z + bv.z, y3 = c3 * rs * wv.w + bv.w;
    // rope: pairs (y0,y1) freq f=2*lane ; (y2,y3) freq f=2*lane+1
    float f0 = (float)(2 * lane), f1 = (float)(2 * lane + 1);
    float a0 = (float)s * exp2f(-f0 * l2b);
    float a1 = (float)s * exp2f(-f1 * l2b);
    float cA = cosf(a0), sA = sinf(a0), cB = cosf(a1), sB = sinf(a1);
    v4bf o;
    o[0] = (bf16)(y0 * cA - y1 * sA);
    o[1] = (bf16)(y0 * sA + y1 * cA);
    o[2] = (bf16)(y2 * cB - y3 * sB);
    o[3] = (bf16)(y2 * sB + y3 * cB);
    *(v4bf*)(dst + d0) = o;
  }
}

// ---------------- causal flash attention (GQA) ----------------
// grid = B*H*(S/128); 8 waves; wave owns 16 q-rows x full D.
__global__ __launch_bounds__(256)
void ga_attn(const bf16* __restrict__ Q,   // [B,H,S,D]
             const bf16* __restrict__ Kt,  // [B,KV,S,D]
             const bf16* __restrict__ Vt,  // [B,KV,D,S]
             float* __restrict__ O)        // [B,H,S,D]
{
  const int nq = Ss / 128;
  const int qbk = blockIdx.x % nq;
  const int bh  = blockIdx.x / nq;
  const int h = bh % H, b = bh / H;
  const int kvh = h / (H / KV);

  const bf16* Qp = Q  + (((size_t)b * H  + h)   * Ss + qbk * 128) * D;
  const bf16* Kp = Kt + (((size_t)b * KV + kvh) * Ss) * D;
  const bf16* Vp = Vt + (((size_t)b * KV + kvh) * (size_t)D) * Ss;

  const int wave = threadIdx.x >> 5, lane = threadIdx.x & 31;
  const int r = lane & 15, sel = lane >> 4, koff = sel * 8;

  __shared__ bf16 Pbuf[8 * 16 * 128];       // per-wave 16x128 P tile (bf16)
  bf16* Pw = Pbuf + wave * 16 * 128;

  // Q fragments: 4 chunks of K-dim (D) of 32
  v16bf qf[4];
  #pragma unroll
  for (int c = 0; c < 4; ++c)
    qf[c] = load_frag(Qp + (size_t)(wave * 16 + r) * D + c * 32 + koff);

  float m_i[8], l_i[8];
  v8f o[8] = {};
  #pragma unroll
  for (int g = 0; g < 8; ++g) { m_i[g] = -1e30f; l_i[g] = 0.f; }

  const float sc = 0.08838834764831845f;    // 1/sqrt(128)
  const int qrow0 = qbk * 128 + wave * 16 + sel * 8;

  for (int j = 0; j <= qbk; ++j) {
    const bf16* Kj = Kp + (size_t)j * 128 * D;
    v8f sacc[8] = {};
    #pragma unroll
    for (int nt = 0; nt < 8; ++nt) {
      v16bf kfv[4];
      #pragma unroll
      for (int c = 0; c < 4; ++c)
        kfv[c] = load_frag(Kj + (size_t)(nt * 16 + r) * D + c * 32 + koff);
      #pragma unroll
      for (int c = 0; c < 4; ++c)
        sacc[nt] = __builtin_amdgcn_wmma_f32_16x16x32_bf16(
            false, qf[c], false, kfv[c], (short)0, sacc[nt], false, false);
    }
    // scale + causal mask
    #pragma unroll
    for (int nt = 0; nt < 8; ++nt)
      #pragma unroll
      for (int g = 0; g < 8; ++g) {
        float v = sacc[nt][g] * sc;
        const int ki = j * 128 + nt * 16 + r;
        if (ki > qrow0 + g) v = -1e30f;
        sacc[nt][g] = v;
      }
    // online softmax update (rows live in 16-lane half-groups)
    #pragma unroll
    for (int g = 0; g < 8; ++g) {
      float mx = sacc[0][g];
      #pragma unroll
      for (int nt = 1; nt < 8; ++nt) mx = fmaxf(mx, sacc[nt][g]);
      mx = grp16_max(mx);
      float mnew = fmaxf(m_i[g], mx);
      float corr = __expf(m_i[g] - mnew);
      m_i[g] = mnew;
      l_i[g] *= corr;
      #pragma unroll
      for (int dt = 0; dt < 8; ++dt) o[dt][g] *= corr;
    }
    // P = exp(S-m): accumulate row sums, spill to LDS in A-fragment-friendly order
    #pragma unroll
    for (int g = 0; g < 8; ++g) {
      float ls = 0.f;
      #pragma unroll
      for (int nt = 0; nt < 8; ++nt) {
        float pv = __expf(sacc[nt][g] - m_i[g]);
        ls += pv;
        Pw[(g + sel * 8) * 128 + nt * 16 + r] = (bf16)pv;
      }
      l_i[g] += grp16_sum(ls);
    }
    // O += P * V   (A from LDS, B from V^T rows)
    #pragma unroll
    for (int dt = 0; dt < 8; ++dt) {
      #pragma unroll
      for (int c = 0; c < 4; ++c) {
        v16bf pa = load_frag(Pw + r * 128 + c * 32 + koff);
        v16bf vb = load_frag(Vp + (size_t)(dt * 16 + r) * Ss + j * 128 + c * 32 + koff);
        o[dt] = __builtin_amdgcn_wmma_f32_16x16x32_bf16(
            false, pa, false, vb, (short)0, o[dt], false, false);
      }
    }
  }
  // epilogue: normalize and store f32 [B,H,S,D]
  float* Op = O + (((size_t)b * H + h) * Ss + qbk * 128 + wave * 16) * D;
  #pragma unroll
  for (int dt = 0; dt < 8; ++dt)
    #pragma unroll
    for (int g = 0; g < 8; ++g)
      Op[(size_t)(sel * 8 + g) * D + dt * 16 + r] = o[dt][g] / l_i[g];
}

// ---------------- gating: A = attn * sigmoid(gate), bf16 [M, HID] ----------------
__global__ void ga_gate(const float* __restrict__ qkv, const float* __restrict__ attn,
                        bf16* __restrict__ Aout) {
  size_t i = (size_t)blockIdx.x * blockDim.x + threadIdx.x;
  if (i >= (size_t)Mrows * HID) return;
  const int m = (int)(i / HID), c = (int)(i % HID);
  const int b = m / Ss, s = m % Ss;
  const int hh = c / D, d = c % D;
  float g = qkv[(size_t)m * NQKV + HID + c];        // gate = qg cols [2048,4096)
  float a = attn[(((size_t)b * H + hh) * Ss + s) * D + d];
  Aout[i] = (bf16)(a * (1.0f / (1.0f + __expf(-g))));
}

// ---------------- host launch ----------------
extern "C" void kernel_launch(void* const* d_in, const int* in_sizes, int n_in,
                              void* d_out, int out_size, void* d_ws, size_t ws_size,
                              hipStream_t stream) {
  const float* x    = (const float*)d_in[0];
  const float* wqkv = (const float*)d_in[1];
  const float* wo   = (const float*)d_in[2];
  const float* qnw  = (const float*)d_in[3];
  const float* qnb  = (const float*)d_in[4];
  const float* knw  = (const float*)d_in[5];
  const float* knb  = (const float*)d_in[6];

  char* p = (char*)d_ws;
  auto take = [&](size_t bytes) {
    char* r = p; p += (bytes + 255) & ~(size_t)255; return r;
  };
  bf16*  x_bf  = (bf16*) take((size_t)Mrows * HID * 2);
  bf16*  wq_bf = (bf16*) take((size_t)NQKV * HID * 2);
  bf16*  wo_bf = (bf16*) take((size_t)HID * HID * 2);
  float* qkvf  = (float*)take((size_t)Mrows * NQKV * 4);
  bf16*  q_bf  = (bf16*) take((size_t)Bb * H  * Ss * D * 2);
  bf16*  k_bf  = (bf16*) take((size_t)Bb * KV * Ss * D * 2);
  bf16*  vT_bf = (bf16*) take((size_t)Bb * KV * D  * Ss * 2);
  float* attnf = (float*)take((size_t)Bb * H  * Ss * D * 4);
  bf16*  a_bf  = (bf16*) take((size_t)Mrows * HID * 2);

  auto cvtGrid = [](size_t n) { return dim3((unsigned)((n / 4 + 255) / 256)); };
  ga_cvt_bf16<<<cvtGrid((size_t)Mrows * HID), 256, 0, stream>>>(x, x_bf, (size_t)Mrows * HID);
  ga_cvt_bf16<<<cvtGrid((size_t)NQKV * HID), 256, 0, stream>>>(wqkv, wq_bf, (size_t)NQKV * HID);
  ga_cvt_bf16<<<cvtGrid((size_t)HID * HID), 256, 0, stream>>>(wo, wo_bf, (size_t)HID * HID);

  ga_gemm_bf16<<<dim3((Mrows / TM) * (NQKV / TN)), 256, 0, stream>>>(
      x_bf, wq_bf, qkvf, Mrows, NQKV, HID);

  ga_normrope<<<dim3(Mrows), 256, 0, stream>>>(qkvf, qnw, qnb, knw, knb,
                                               q_bf, k_bf, vT_bf);

  ga_attn<<<dim3(Bb * H * (Ss / 128)), 256, 0, stream>>>(q_bf, k_bf, vT_bf, attnf);

  ga_gate<<<dim3((unsigned)(((size_t)Mrows * HID + 255) / 256)), 256, 0, stream>>>(
      qkvf, attnf, a_bf);

  ga_gemm_bf16<<<dim3((Mrows / TM) * (HID / TN)), 256, 0, stream>>>(
      a_bf, wo_bf, (float*)d_out, Mrows, HID, HID);
}
